// DAWNBlock_31035433681149
// MI455X (gfx1250) — compile-verified, matrix operand order checked by reading, not run on hardware
//
#include <hip/hip_runtime.h>
#include <hip/hip_bf16.h>
#include <math.h>

typedef __attribute__((ext_vector_type(16))) _Float16 v16h;
typedef __attribute__((ext_vector_type(8)))  _Float16 v8h;
typedef __attribute__((ext_vector_type(8)))  float    v8f;
typedef __attribute__((ext_vector_type(4)))  unsigned int u32x4;
typedef __attribute__((ext_vector_type(8)))  int       i32x8;
typedef __attribute__((ext_vector_type(4)))  int       i32x4;

#define S_LEN 2048
#define D_IN  2048
#define DS    64
#define NB    8
#define NCOMB 2048           // combined neuron count (1536 + 512 rel_k)
#define NTOK  (NB * S_LEN)   // 16384 tokens

// ---------------------------------------------------------------- fragments

// 16x32 f16 fragment from an f16 source (row-major, stride halves).
// Used for A (rows = M) and B (rows = N, assuming symmetric lane layout).
__device__ inline v16h frag_f16(const _Float16* base, int row0, int stride,
                                int k0, int lane) {
    int r    = row0 + (lane & 15);
    int koff = (lane >> 4) << 3;
    const _Float16* p = base + (size_t)r * stride + k0 + koff;
    v8h lo = *(const v8h*)p;
    v8h hi = *(const v8h*)(p + 16);
    v16h out;
#pragma unroll
    for (int i = 0; i < 8; ++i) { out[i] = lo[i]; out[i + 8] = hi[i]; }
    return out;
}

// 16x32 A fragment from an f32 LDS tile (row stride 64 floats).
__device__ inline v16h frag_a_lds(const float* base, int sub, int lane) {
    const float* p = base + (size_t)(lane & 15) * 64 + sub + ((lane >> 4) << 3);
    v16h a;
#pragma unroll
    for (int i = 0; i < 8; ++i) {
        a[i]     = (_Float16)p[i];
        a[i + 8] = (_Float16)p[16 + i];
    }
    return a;
}

__device__ inline float wave_max(float v) {
#pragma unroll
    for (int m = 16; m >= 1; m >>= 1) v = fmaxf(v, __shfl_xor(v, m, 32));
    return v;
}
__device__ inline float wave_sum(float v) {
#pragma unroll
    for (int m = 16; m >= 1; m >>= 1) v += __shfl_xor(v, m, 32);
    return v;
}

// ---------------------------------------------------------------- kernel 0a
// One wave per embedding row: L2-normalize into f16 combined table E16.
__global__ __launch_bounds__(256) void prep_norm_kernel(
    const float* __restrict__ neuron_emb,      // [2560,64] (first 1536 used)
    const float* __restrict__ rel_k,           // [512,64]
    _Float16* __restrict__ E16)                // [2048,64]
{
    int wave = blockIdx.x * 8 + (threadIdx.x >> 5);
    int lane = threadIdx.x & 31;
    if (wave >= NCOMB) return;
    const float* src = (wave < 1536) ? (neuron_emb + (size_t)wave * DS)
                                     : (rel_k + (size_t)(wave - 1536) * DS);
    float e0 = src[lane], e1 = src[lane + 32];
    float ss = wave_sum(e0 * e0 + e1 * e1);
    float inv = rsqrtf(ss);
    E16[(size_t)wave * DS + lane]      = (_Float16)(e0 * inv);
    E16[(size_t)wave * DS + lane + 32] = (_Float16)(e1 * inv);
}

// ---------------------------------------------------------------- kernel 0b
// Bonus vector, proj_w f32->f16, zero the w accumulator.
__global__ __launch_bounds__(256) void prep_misc_kernel(
    const float* __restrict__ proj_w,          // [64,2048]
    const float* __restrict__ ema_qk, const float* __restrict__ ema_v,
    const float* __restrict__ ema_rel, const float* __restrict__ ema_val,
    _Float16* __restrict__ pw16,               // [64,2048]
    float* __restrict__ bonus,                 // [2048]
    float* __restrict__ w_acc)                 // [8,2048]
{
    int i = blockIdx.x * blockDim.x + threadIdx.x;
    if (i < DS * D_IN) pw16[i] = (_Float16)proj_w[i];
    if (i < NB * NCOMB) w_acc[i] = 0.0f;
    if (i < NCOMB) {
        float ema;
        if      (i < 512)  ema = ema_qk[i];
        else if (i < 768)  ema = ema_v[i - 512];
        else if (i < 1280) ema = ema_rel[i - 768];
        else if (i < 1536) ema = ema_val[i - 1280];
        else               ema = ema_rel[i - 1536];
        float e = 1.0f - ema / 1.5f;
        bonus[i] = fminf(fmaxf(e, 0.0f), 1.0f);   // * EXC_W(=1)
    }
}

// ---------------------------------------------------------------- kernel 1
// h = x @ proj_w^T  (M=16384, K=2048, N=64), f16 WMMA with f32 accumulate.
// 4 waves per block; each wave owns 16 tokens x all 64 output columns, so
// x is consumed exactly once.  x tiles (64 tokens x 64 floats) are streamed
// into LDS by the Tensor Data Mover, double buffered, waited on TENSORcnt.
__global__ __launch_bounds__(128) void gemm_h_tdm_kernel(
    const float* __restrict__ x,               // [16384,2048]
    const _Float16* __restrict__ pw16,         // [64,2048]
    const float* __restrict__ proj_b,          // [64]
    _Float16* __restrict__ h16)                // [16384,64]
{
    __shared__ float xbuf[2][64][64];          // 32 KB, double buffered
    int lane = threadIdx.x & 31;
    int wv   = threadIdx.x >> 5;               // 0..3
    int t0   = blockIdx.x * 64;

    unsigned lds_base  = (unsigned)(uintptr_t)(&xbuf[0][0][0]);
    const unsigned buf_bytes = 64 * 64 * 4;

    v8f acc[4] = {};

    auto issue_tdm = [&](int chunk, int buf) {
        unsigned long long ga =
            (unsigned long long)(uintptr_t)(x + (size_t)t0 * D_IN + chunk * 64);
        u32x4 g0;
        g0[0] = 1u;                                        // count = 1 (valid D#)
        g0[1] = lds_base + (unsigned)buf * buf_bytes;      // lds_addr
        g0[2] = (unsigned)(ga & 0xFFFFFFFFu);              // global_addr[31:0]
        g0[3] = (unsigned)((ga >> 32) & 0x1FFFFFFu)        // global_addr[56:32]
              | (2u << 30);                                // type = 2 (image)
        i32x8 g1;
        g1[0] = (int)(2u << 16);                           // data_size = 4 bytes
        g1[1] = (int)((unsigned)(D_IN & 0xFFFF) << 16);    // tensor_dim0 lo16
        g1[2] = (int)(((unsigned)D_IN >> 16)               // tensor_dim0 hi16
              | ((unsigned)(NTOK & 0xFFFF) << 16));        // tensor_dim1 lo16
        g1[3] = (int)(((unsigned)NTOK >> 16)               // tensor_dim1 hi16
              | (64u << 16));                              // tile_dim0 = 64
        g1[4] = (int)64u;                                  // tile_dim1 = 64, tile_dim2 = 0
        g1[5] = (int)(unsigned)D_IN;                       // tensor_dim0_stride lo32
        g1[6] = 0;                                         // stride hi, dim1_stride lo
        g1[7] = 0;
        i32x4 z4 = {0, 0, 0, 0};                           // groups 2/3 unused (2D)
        i32x8 z8 = {0, 0, 0, 0, 0, 0, 0, 0};
        __builtin_amdgcn_tensor_load_to_lds(g0, g1, z4, z4, z8, 0);
    };

    if (wv == 0) {
        issue_tdm(0, 0);
        __builtin_amdgcn_s_wait_tensorcnt(0);
    }
    __syncthreads();

    for (int c = 0; c < D_IN / 64; ++c) {                  // 32 chunks of K=64
        int cur = c & 1;
        if (wv == 0 && c + 1 < D_IN / 64) issue_tdm(c + 1, cur ^ 1);
#pragma unroll
        for (int sub = 0; sub < 64; sub += 32) {
            v16h a = frag_a_lds(&xbuf[cur][wv * 16][0], sub, lane);
#pragma unroll
            for (int nt = 0; nt < 4; ++nt) {
                v16h b = frag_f16(pw16, nt * 16, D_IN, c * 64 + sub, lane);
                acc[nt] = __builtin_amdgcn_wmma_f32_16x16x32_f16(
                              false, a, false, b, (short)0, acc[nt],
                              false, false);
            }
        }
        if (wv == 0) __builtin_amdgcn_s_wait_tensorcnt(0);
        __syncthreads();
    }

#pragma unroll
    for (int nt = 0; nt < 4; ++nt) {
        float bias = proj_b[nt * 16 + (lane & 15)];
#pragma unroll
        for (int r = 0; r < 8; ++r) {
            int row = t0 + wv * 16 + r + ((lane >> 4) << 3);
            h16[(size_t)row * DS + nt * 16 + (lane & 15)] =
                (_Float16)(acc[nt][r] + bias);
        }
    }
}

// ---------------------------------------------------------------- kernel 2
// Per 16-token tile: logits (WMMA) -> segment softmax -> importance-weighted
// reduction into w_acc.  512 threads = 16 waves; 64 KB LDS logits tile.
__global__ __launch_bounds__(512) void logits_route_kernel(
    const _Float16* __restrict__ h16,          // [16384,64]
    const _Float16* __restrict__ E16,          // [2048,64]
    const float* __restrict__ bonus,           // [2048]
    const float* __restrict__ importance,      // [8,2048]
    float* __restrict__ w_acc)                 // [8,2048]
{
    __shared__ _Float16 slog[16 * NCOMB];      // 65536 bytes
    int lane = threadIdx.x & 31;
    int wv   = threadIdx.x >> 5;               // 0..15
    int t0   = blockIdx.x * 16;                // tokens share a batch (S%16==0)
    int b    = t0 >> 11;

    // ---- stage 1: logits = h @ E^T + bonus, 8 neuron tiles per wave
    v16h a0 = frag_f16(h16, t0, DS, 0,  lane);
    v16h a1 = frag_f16(h16, t0, DS, 32, lane);
#pragma unroll
    for (int jj = 0; jj < 8; ++jj) {
        int n0 = (wv + 16 * jj) * 16;
        v16h b0 = frag_f16(E16, n0, DS, 0,  lane);
        v16h b1 = frag_f16(E16, n0, DS, 32, lane);
        v8f c = {};
        c = __builtin_amdgcn_wmma_f32_16x16x32_f16(false, a0, false, b0,
                                                   (short)0, c, false, false);
        c = __builtin_amdgcn_wmma_f32_16x16x32_f16(false, a1, false, b1,
                                                   (short)0, c, false, false);
        int n  = n0 + (lane & 15);
        float bn = bonus[n];
#pragma unroll
        for (int r = 0; r < 8; ++r) {
            int m = r + ((lane >> 4) << 3);
            slog[m * NCOMB + n] = (_Float16)(c[r] + bn);
        }
    }
    __syncthreads();

    // ---- stage 2: wave wv owns token wv; per-segment softmax, scale by imp
    {
        int t = wv;
        int s = (t0 & (S_LEN - 1)) + t;
        float imp = importance[(size_t)b * S_LEN + s];
        const int segoff[5] = {0, 512, 768, 1280, 1536};
        const int seglen[5] = {512, 256, 512, 256, 512};
        _Float16* row = &slog[t * NCOMB];
#pragma unroll
        for (int sg = 0; sg < 5; ++sg) {
            int off = segoff[sg], len = seglen[sg];
            float m = -3.4e38f;
            for (int i = lane; i < len; i += 32)
                m = fmaxf(m, (float)row[off + i]);
            m = wave_max(m);
            float ssum = 0.0f;
            for (int i = lane; i < len; i += 32)
                ssum += __expf((float)row[off + i] - m);
            ssum = wave_sum(ssum);
            float inv = imp / ssum;
            for (int i = lane; i < len; i += 32)
                row[off + i] = (_Float16)(__expf((float)row[off + i] - m) * inv);
        }
    }
    __syncthreads();

    // ---- stage 3: column-sum over the 16 tokens, one atomic per neuron
    for (int n = threadIdx.x; n < NCOMB; n += 512) {
        float acc = 0.0f;
#pragma unroll
        for (int t = 0; t < 16; ++t) acc += (float)slog[t * NCOMB + n];
        atomicAdd(&w_acc[(size_t)b * NCOMB + n], acc);
    }
}

// ---------------------------------------------------------------- kernel 3
// 40 blocks: (batch, route) -> top-k, softmax(top values), sorted indices.
__global__ __launch_bounds__(256) void topk_kernel(
    const float* __restrict__ w_acc,           // [8,2048]
    float* __restrict__ out)                   // [4096] floats then int idx
{
    __shared__ float vals[512];
    __shared__ float red_v[256];
    __shared__ int   red_i[256];
    __shared__ float tw[64];
    __shared__ int   tidx[64];

    const int offs[5] = {0, 512, 768, 1536, 1280};   // slice in combined w
    const int lens[5] = {512, 256, 512, 512, 256};
    const int ks[5]   = {64, 32, 64, 64, 32};
    const int soff[5] = {0, 64, 96, 160, 224};       // into soft_all row
    const int ioff[5] = {2048, 2560, 2816, 3328, 3840};

    int task = blockIdx.x, b = task / 5, rt = task % 5;
    int tid  = threadIdx.x;
    int len  = lens[rt], k = ks[rt];

    for (int i = tid; i < 512; i += 256)
        vals[i] = (i < len) ? w_acc[(size_t)b * NCOMB + offs[rt] + i] : -3.4e38f;
    __syncthreads();

    for (int j = 0; j < k; ++j) {
        float v0 = vals[tid], v1 = vals[tid + 256];
        float v; int idx;
        if (v1 > v0) { v = v1; idx = tid + 256; } else { v = v0; idx = tid; }
        red_v[tid] = v; red_i[tid] = idx;
        __syncthreads();
        for (int s = 128; s > 0; s >>= 1) {
            if (tid < s && red_v[tid + s] > red_v[tid]) {
                red_v[tid] = red_v[tid + s]; red_i[tid] = red_i[tid + s];
            }
            __syncthreads();
        }
        if (tid == 0) {
            tw[j] = red_v[0]; tidx[j] = red_i[0];
            vals[red_i[0]] = -3.4e38f;
        }
        __syncthreads();
    }

    if (tid == 0) {                                  // softmax over top values
        float m = tw[0];
        for (int j = 1; j < k; ++j) m = fmaxf(m, tw[j]);
        float s = 0.0f;
        for (int j = 0; j < k; ++j) s += __expf(tw[j] - m);
        float inv = 1.0f / s;
        for (int j = 0; j < k; ++j)
            out[b * 256 + soff[rt] + j] = __expf(tw[j] - m) * inv;
    }
    __syncthreads();
    if (tid < k) {                                   // rank-sort indices
        int my = tidx[tid], rank = 0;
        for (int j = 0; j < k; ++j) rank += (tidx[j] < my);
        ((int*)out)[ioff[rt] + b * k + rank] = my;
    }
}

// ---------------------------------------------------------------- launch

extern "C" void kernel_launch(void* const* d_in, const int* in_sizes, int n_in,
                              void* d_out, int out_size, void* d_ws, size_t ws_size,
                              hipStream_t stream) {
    const float* x          = (const float*)d_in[0];
    const float* importance = (const float*)d_in[1];
    const float* proj_w     = (const float*)d_in[2];
    const float* proj_b     = (const float*)d_in[3];
    const float* neuron_emb = (const float*)d_in[4];
    const float* rel_k      = (const float*)d_in[5];
    const float* ema_qk     = (const float*)d_in[6];
    const float* ema_v      = (const float*)d_in[7];
    const float* ema_rel    = (const float*)d_in[8];
    const float* ema_val    = (const float*)d_in[9];
    float* out = (float*)d_out;

    // workspace layout (256B-aligned regions)
    char* ws = (char*)d_ws;
    _Float16* E16   = (_Float16*)(ws + 0);          // 2048*64*2 = 262144
    _Float16* pw16  = (_Float16*)(ws + 262144);     // 64*2048*2 = 262144
    float*    bonus = (float*)   (ws + 524288);     // 2048*4    = 8192
    float*    w_acc = (float*)   (ws + 532480);     // 8*2048*4  = 65536
    _Float16* h16   = (_Float16*)(ws + 598016);     // 16384*64*2 = 2097152

    prep_norm_kernel<<<256, 256, 0, stream>>>(neuron_emb, rel_k, E16);
    prep_misc_kernel<<<512, 256, 0, stream>>>(proj_w, ema_qk, ema_v, ema_rel,
                                              ema_val, pw16, bonus, w_acc);
    gemm_h_tdm_kernel<<<256, 128, 0, stream>>>(x, pw16, proj_b, h16);
    logits_route_kernel<<<1024, 512, 0, stream>>>(h16, E16, bonus,
                                                  importance, w_acc);
    topk_kernel<<<40, 256, 0, stream>>>(w_acc, out);
}
